// EMG_LSTM_58351425683693
// MI455X (gfx1250) — compile-verified, matrix operand order checked by reading, not run on hardware
//
#include <hip/hip_runtime.h>

// ---------------------------------------------------------------------------
// Fused LSTM (B=4096, T=512, IN=16, H=50) + dense(20) + softmax for gfx1250.
// One block = 16 batch rows, 128 threads (4 wave32). All weights + recurrent
// state live in LDS for the whole sequence. Gate pre-activations computed with
// V_WMMA_F32_16X16X32_F16 (f16 inputs, f32 accumulate).
//
// Round-3 changes (critical-path latency):
//  * gate-math loop fully unrolled as 16 rows x 8 lanes x 7 k-slices:
//    no integer div/mod magic, no exec-mask loop exit, constant LDS offsets
//  * x-staging loop unrolled (2 fixed iterations)
// ---------------------------------------------------------------------------

#define HID     50
#define HID4    200
#define INDIM   16
#define TSTEPS  512
#define NCLS    20
#define BTILE   16
#define NTILES  13          // ceil(208/16): 200 gate cols padded to 208
#define NWAVES  4

typedef __attribute__((ext_vector_type(16))) _Float16 v16h;
typedef __attribute__((ext_vector_type(8)))  float    v8f;

struct __align__(32) Smem {
    _Float16 Bih[NTILES][32][16];       // W_ih as B-fragments, K padded 16->32
    _Float16 Bhh[NTILES][2][32][16];    // W_hh as B-fragments, K padded 50->64
    _Float16 xA[2][32][16];             // x_t tile A-fragment, double-buffered
    _Float16 hA[2][32][16];             // h_t tile as A-fragments (K pad -> 64)
    float    bias[208];                 // b_ih + b_hh, padded with zeros
    float    c[BTILE][HID];             // cell state, f32
    float    gates[BTILE][208];         // gate pre-activations / logits scratch
};

#define LOG2E 1.4426950408889634f

__device__ __forceinline__ float fexp(float v)  { return __builtin_amdgcn_exp2f(v * LOG2E); }
__device__ __forceinline__ float frcp(float v)  { return __builtin_amdgcn_rcpf(v); }
__device__ __forceinline__ float sigm(float v)  { return frcp(1.0f + fexp(-v)); }
__device__ __forceinline__ float ftanh(float v) { return 1.0f - 2.0f * frcp(1.0f + fexp(2.0f * v)); }

// A-fragment (f16 16x32) lane/element mapping for value (row m, k):
//   group = k/8; lane = m + 16*(group&1); elem = (k&7) + 8*(group>>1)
// B-fragment (f16 32x16) mirrors with m -> n.

__global__ __launch_bounds__(128)
void lstm_fused(const float* __restrict__ x,
                const float* __restrict__ W_ih,
                const float* __restrict__ W_hh,
                const float* __restrict__ b_ih,
                const float* __restrict__ b_hh,
                const float* __restrict__ W_dense,
                const float* __restrict__ b_dense,
                float* __restrict__ out)
{
    __shared__ Smem s;
    const int tid  = threadIdx.x;
    const int lane = tid & 31;
    // wave index is uniform within a wave: force it scalar for SALU loop ctrl
    const int wave = __builtin_amdgcn_readfirstlane(tid >> 5);
    const int b0   = blockIdx.x * BTILE;

    // ---------------- one-time LDS setup: swizzle weights into fragments ----
    for (int idx = tid; idx < NTILES * 32 * 16; idx += 128) {
        int e  = idx & 15;
        int l  = (idx >> 4) & 31;
        int nt = idx >> 9;
        int k   = (e & 7) + 8 * (((e >> 3) << 1) | (l >> 4));
        int col = nt * 16 + (l & 15);
        float v = (k < INDIM && col < HID4) ? W_ih[col * INDIM + k] : 0.0f;
        s.Bih[nt][l][e] = (_Float16)v;
    }
    for (int idx = tid; idx < NTILES * 2 * 32 * 16; idx += 128) {
        int e  = idx & 15;
        int l  = (idx >> 4) & 31;
        int ch = (idx >> 9) & 1;
        int nt = idx >> 10;
        int kk  = (e & 7) + 8 * (((e >> 3) << 1) | (l >> 4));
        int k   = ch * 32 + kk;
        int col = nt * 16 + (l & 15);
        float v = (k < HID && col < HID4) ? W_hh[col * HID + k] : 0.0f;
        s.Bhh[nt][ch][l][e] = (_Float16)v;
    }
    for (int j = tid; j < 208; j += 128)
        s.bias[j] = (j < HID4) ? (b_ih[j] + b_hh[j]) : 0.0f;
    for (int idx = tid; idx < BTILE * HID; idx += 128)
        (&s.c[0][0])[idx] = 0.0f;
    for (int idx = tid; idx < 2 * 32 * 16; idx += 128)
        (&s.hA[0][0][0])[idx] = (_Float16)0.0f;
    for (int idx = tid; idx < 2 * 32 * 16; idx += 128)
        (&s.xA[0][0][0])[idx] = (_Float16)0.0f;  // zero both buffers (K pad)
    // Pre-stage x_0 into buffer 0.
    {
        int m = tid >> 4, k = tid & 15;
        if (tid < BTILE * INDIM) {      // 128 threads stage first half
            float xv = x[((size_t)(b0 + m)) * TSTEPS * INDIM + k];
            s.xA[0][m + ((k >> 3) ? 16 : 0)][k & 7] = (_Float16)xv;
        }
        int idx2 = tid + 128;           // second half
        m = idx2 >> 4; k = idx2 & 15;
        float xv = x[((size_t)(b0 + m)) * TSTEPS * INDIM + k];
        s.xA[0][m + ((k >> 3) ? 16 : 0)][k & 7] = (_Float16)xv;
    }
    __syncthreads();

    // ---------------- sequential recurrence over T -------------------------
    const int hi  = lane >> 4;
    const int nn  = lane & 15;
    const int gm  = tid >> 3;           // gate-math: row 0..15
    const int gk0 = tid & 7;            // gate-math: base k (slice stride 8)
    for (int t = 0; t < TSTEPS; ++t) {
        const int cur = t & 1;

        // Gate pre-activations: 13 column tiles split across the 4 waves.
        v16h a_x  = *(const v16h*)(&s.xA[cur][lane][0]);
        v16h a_h0 = *(const v16h*)(&s.hA[0][lane][0]);
        v16h a_h1 = *(const v16h*)(&s.hA[1][lane][0]);
        for (int nt = wave; nt < NTILES; nt += NWAVES) {
            v16h bi  = *(const v16h*)(&s.Bih[nt][lane][0]);
            v16h bh0 = *(const v16h*)(&s.Bhh[nt][0][lane][0]);
            v16h bh1 = *(const v16h*)(&s.Bhh[nt][1][lane][0]);
            v8f acc = {};
            acc = __builtin_amdgcn_wmma_f32_16x16x32_f16(false, a_x,  false, bi,
                                                         (short)0, acc, false, false);
            acc = __builtin_amdgcn_wmma_f32_16x16x32_f16(false, a_h0, false, bh0,
                                                         (short)0, acc, false, false);
            acc = __builtin_amdgcn_wmma_f32_16x16x32_f16(false, a_h1, false, bh1,
                                                         (short)0, acc, false, false);
            int col = nt * 16 + nn;
            float bsum = s.bias[col];       // zero in the padded columns
            #pragma unroll
            for (int r = 0; r < 8; ++r) {
                s.gates[r + hi * 8][col] = acc[r] + bsum;   // D: M = r + 8*hi, N = nn
            }
        }
        __syncthreads();

        // Cell update (i,f,g,o order) + repack h into next step's A-fragments.
        // 800 elements = 16 rows x 8 lanes x 7 slices; all offsets fold to
        // constants, only the last slice (k = 48..55) needs predication.
        #pragma unroll
        for (int i = 0; i < 7; ++i) {
            const int k = gk0 + 8 * i;
            if (i < 6 || k < HID) {
                float iv = s.gates[gm][k];
                float fv = s.gates[gm][HID + k];
                float gv = s.gates[gm][2 * HID + k];
                float ov = s.gates[gm][3 * HID + k];
                float cn = sigm(fv) * s.c[gm][k] + sigm(iv) * ftanh(gv);
                s.c[gm][k] = cn;
                float hv = sigm(ov) * ftanh(cn);
                const int ch = k >> 5;          // constant per slice
                const int r  = k & 31;
                const int g  = r >> 3;          // constant per slice
                const int l  = gm + 16 * (g & 1);
                const int e  = (r & 7) + 8 * (g >> 1);
                s.hA[ch][l][e] = (_Float16)hv;  // K=50..63 region stays zero
            }
        }
        // Stage x_{t+1} into the other buffer while gate math runs.
        if (t + 1 < TSTEPS) {
            #pragma unroll
            for (int u = 0; u < 2; ++u) {
                int idx = tid + u * 128;
                int m = idx >> 4;
                int k = idx & 15;
                float xv = x[(((size_t)(b0 + m)) * TSTEPS + (t + 1)) * INDIM + k];
                s.xA[cur ^ 1][m + ((k >> 3) ? 16 : 0)][k & 7] = (_Float16)xv;
            }
            if (t + 2 < TSTEPS && tid < BTILE) {
                __builtin_prefetch(&x[(((size_t)(b0 + tid)) * TSTEPS + (t + 2)) * INDIM], 0, 1);
            }
        }
        __syncthreads();
    }

    // ---------------- head: logits = c @ W_denseᵀ + b, then softmax --------
    for (int idx = tid; idx < BTILE * NCLS; idx += 128) {
        int m = idx / NCLS;
        int j = idx - m * NCLS;
        float acc = b_dense[j];
        for (int k = 0; k < HID; ++k)
            acc += s.c[m][k] * W_dense[j * HID + k];
        s.gates[m][j] = acc;                // reuse gates as logits scratch
    }
    __syncthreads();
    if (tid < BTILE) {
        int m = tid;
        float mx = -1e30f;
        for (int j = 0; j < NCLS; ++j) mx = fmaxf(mx, s.gates[m][j]);
        float ex[NCLS];
        float sum = 0.0f;
        for (int j = 0; j < NCLS; ++j) { ex[j] = fexp(s.gates[m][j] - mx); sum += ex[j]; }
        float inv = frcp(sum);
        for (int j = 0; j < NCLS; ++j)
            out[(size_t)(b0 + m) * NCLS + j] = ex[j] * inv;
    }
}

extern "C" void kernel_launch(void* const* d_in, const int* in_sizes, int n_in,
                              void* d_out, int out_size, void* d_ws, size_t ws_size,
                              hipStream_t stream) {
    const float* x       = (const float*)d_in[0];
    const float* W_ih    = (const float*)d_in[1];
    const float* W_hh    = (const float*)d_in[2];
    const float* b_ih    = (const float*)d_in[3];
    const float* b_hh    = (const float*)d_in[4];
    const float* W_dense = (const float*)d_in[5];
    const float* b_dense = (const float*)d_in[6];
    float* out = (float*)d_out;

    const int B = in_sizes[0] / (TSTEPS * INDIM);   // 4096
    dim3 grid(B / BTILE);                           // 256 blocks
    dim3 block(128);                                // 4 wave32 per block
    lstm_fused<<<grid, block, 0, stream>>>(x, W_ih, W_hh, b_ih, b_hh,
                                           W_dense, b_dense, out);
}
